// GraphSAGE_86019605004681
// MI455X (gfx1250) — matrix operand order repaired and checked by reading, not compiled, not run on hardware
//
#include <hip/hip_runtime.h>

typedef __bf16 bf16;
typedef __attribute__((ext_vector_type(16))) __bf16 v16bf;
typedef __attribute__((ext_vector_type(8)))  __bf16 v8bf;
typedef __attribute__((ext_vector_type(8)))  float  v8f;

#define NF 128

__device__ __forceinline__ void split_bf(float x, bf16& h, bf16& l) {
  h = (bf16)x;
  l = (bf16)(x - (float)h);
}

__device__ __forceinline__ void atomic_add_f32(float* p, float v) {
  __hip_atomic_fetch_add(p, v, __ATOMIC_RELAXED, __HIP_MEMORY_SCOPE_AGENT);
}

// ---------------- degree: deg[dst] += 1 ----------------
__global__ void __launch_bounds__(256) k_deg(const int* __restrict__ dst,
                                             float* __restrict__ deg, int E) {
  int e = blockIdx.x * 256 + threadIdx.x;
  if (e < E) atomic_add_f32(&deg[dst[e]], 1.0f);
}

// ---------------- split 4 weight matrices (128x128 each) into bf16 hi/lo ----------------
__global__ void __launch_bounds__(256) k_wsplit(const float* __restrict__ W0,
                                                const float* __restrict__ W1,
                                                const float* __restrict__ W2,
                                                const float* __restrict__ W3,
                                                bf16* __restrict__ hi, bf16* __restrict__ lo) {
  int idx = blockIdx.x * 256 + threadIdx.x;   // 0 .. 65535
  int mat = idx >> 14;
  int off = idx & 16383;
  const float* W = (mat == 0) ? W0 : (mat == 1) ? W1 : (mat == 2) ? W2 : W3;
  float x = W[off];
  bf16 h, l;
  split_bf(x, h, l);
  hi[idx] = h;
  lo[idx] = l;
}

// ---------------- edge scatter: one wave32 per edge, float4 per lane ----------------
__global__ void __launch_bounds__(256) k_scatter(const float* __restrict__ h,
                                                 const float* __restrict__ w,
                                                 const int* __restrict__ src,
                                                 const int* __restrict__ dst,
                                                 float* __restrict__ s, int E) {
  int gid = blockIdx.x * 256 + threadIdx.x;
  int e = gid >> 5;
  int lane = gid & 31;
  if (e >= E) return;
  int sv = src[e];
  int dv = dst[e];
  float we = w[e];
  const float4 v = *(const float4*)(h + (size_t)sv * NF + lane * 4);
  float* p = s + (size_t)dv * NF + lane * 4;
  atomic_add_f32(p + 0, v.x * we);
  atomic_add_f32(p + 1, v.y * we);
  atomic_add_f32(p + 2, v.z * we);
  atomic_add_f32(p + 3, v.w * we);
}

// ---- A fragment from LDS tile (row-major [16][128]); ISA 16-bit A layout:
//      lane L: m = L%16, elems 0..7 -> k = k0..k0+7, elems 8..15 -> k = k0+16..k0+23
__device__ __forceinline__ v16bf lda(const bf16* base, int m, int k0) {
  const bf16* p = base + m * NF + k0;
  v8bf c0 = *(const v8bf*)p;
  v8bf c1 = *(const v8bf*)(p + 16);
  v16bf r;
#pragma unroll
  for (int i = 0; i < 8; ++i) { r[i] = c0[i]; r[i + 8] = c1[i]; }
  return r;
}

// ---- B fragment: B[k][n] = W[j0+n][k]; lane L: n = L%16, elems 0..7 -> k = k0..k0+7, elems 8..15 -> k = k0+16..k0+23
__device__ __forceinline__ v16bf ldb(const bf16* __restrict__ W, int j, int k0) {
  const bf16* p = W + j * NF + k0;
  v8bf c0 = *(const v8bf*)p;
  v8bf c1 = *(const v8bf*)(p + 16);
  v16bf r;
#pragma unroll
  for (int i = 0; i < 8; ++i) { r[i] = c0[i]; r[i + 8] = c1[i]; }
  return r;
}

#define WMMA_BF16(A, B, C) \
  __builtin_amdgcn_wmma_f32_16x16x32_bf16(false, (A), false, (B), (short)0, (C), false, false)

// ---------------- fused SAGE layer GEMM:
//   out = X @ Ws^T + bias + (S/max(deg,1)) @ Wn^T   (optionally ReLU)
// block = 256 thr = 8 waves; block owns 16 node rows; wave w owns columns [16w,16w+16)
template <bool RELU>
__global__ void __launch_bounds__(256) k_sage_gemm(
    const float* __restrict__ X, const float* __restrict__ S, const float* __restrict__ deg,
    const bf16* __restrict__ Whs, const bf16* __restrict__ Wls,
    const bf16* __restrict__ Whn, const bf16* __restrict__ Wln,
    const float* __restrict__ bias, float* __restrict__ out, int n) {
  __shared__ __attribute__((aligned(16))) bf16 Ahs[16 * NF];
  __shared__ __attribute__((aligned(16))) bf16 Als[16 * NF];
  __shared__ __attribute__((aligned(16))) bf16 Ahn[16 * NF];
  __shared__ __attribute__((aligned(16))) bf16 Aln[16 * NF];

  const int node0 = blockIdx.x * 16;

  // stage A tiles into LDS as bf16 hi/lo (8 elements per thread per matrix)
  {
    int base = threadIdx.x * 8;            // 0..2040
    int row = base >> 7;                   // 0..15
    int col = base & 127;
    int grow = node0 + row;
    if (grow > n - 1) grow = n - 1;
    const float4 x0 = *(const float4*)(X + (size_t)grow * NF + col);
    const float4 x1 = *(const float4*)(X + (size_t)grow * NF + col + 4);
    const float4 s0 = *(const float4*)(S + (size_t)grow * NF + col);
    const float4 s1 = *(const float4*)(S + (size_t)grow * NF + col + 4);
    float inv = 1.0f / fmaxf(deg[grow], 1.0f);
    float xf[8] = {x0.x, x0.y, x0.z, x0.w, x1.x, x1.y, x1.z, x1.w};
    float sf[8] = {s0.x * inv, s0.y * inv, s0.z * inv, s0.w * inv,
                   s1.x * inv, s1.y * inv, s1.z * inv, s1.w * inv};
    v8bf hx, lx, hs, ls;
#pragma unroll
    for (int i = 0; i < 8; ++i) {
      bf16 h, l;
      split_bf(xf[i], h, l); hx[i] = h; lx[i] = l;
      split_bf(sf[i], h, l); hs[i] = h; ls[i] = l;
    }
    *(v8bf*)(Ahs + base) = hx;
    *(v8bf*)(Als + base) = lx;
    *(v8bf*)(Ahn + base) = hs;
    *(v8bf*)(Aln + base) = ls;
  }
  __syncthreads();

  const int wave = threadIdx.x >> 5;
  const int lane = threadIdx.x & 31;
  const int jr   = lane & 15;   // A row m / B column n / output column within tile
  const int half = lane >> 4;
  const int j0   = wave * 16;
  const int jrow = j0 + jr;     // weight-matrix row / output feature index

  v8f acc = {};
#pragma unroll
  for (int s = 0; s < 4; ++s) {        // K = 128 in 4 slices of 32
    const int kA = s * 32 + half * 8;
    const int kB = s * 32 + half * 8;
    v16bf ahx = lda(Ahs, jr, kA);
    v16bf alx = lda(Als, jr, kA);
    v16bf ahn = lda(Ahn, jr, kA);
    v16bf aln = lda(Aln, jr, kA);
    v16bf bhs = ldb(Whs, jrow, kB);
    v16bf bls = ldb(Wls, jrow, kB);
    v16bf bhn = ldb(Whn, jrow, kB);
    v16bf bln = ldb(Wln, jrow, kB);
    // split-precision: hi*hi + hi*lo + lo*hi (fp32 accumulate)
    acc = WMMA_BF16(ahx, bhs, acc);
    acc = WMMA_BF16(ahx, bls, acc);
    acc = WMMA_BF16(alx, bhs, acc);
    acc = WMMA_BF16(ahn, bhn, acc);
    acc = WMMA_BF16(ahn, bln, acc);
    acc = WMMA_BF16(aln, bhn, acc);
  }

  const float b = bias[jrow];
  const int row0 = node0 + half * 8;   // C/D layout: m = r + 8*half
#pragma unroll
  for (int r = 0; r < 8; ++r) {
    float v = acc[r] + b;
    if (RELU) v = fmaxf(v, 0.0f);
    int row = row0 + r;
    if (row < n) out[(size_t)row * NF + jrow] = v;
  }
}

extern "C" void kernel_launch(void* const* d_in, const int* in_sizes, int n_in,
                              void* d_out, int out_size, void* d_ws, size_t ws_size,
                              hipStream_t stream) {
  const float* in_feat  = (const float*)d_in[0];
  const float* weights  = (const float*)d_in[1];
  const int*   src      = (const int*)d_in[2];
  const int*   dst      = (const int*)d_in[3];
  const float* W_self1  = (const float*)d_in[4];
  const float* b_self1  = (const float*)d_in[5];
  const float* W_neigh1 = (const float*)d_in[6];
  const float* W_self2  = (const float*)d_in[7];
  const float* b_self2  = (const float*)d_in[8];
  const float* W_neigh2 = (const float*)d_in[9];
  float* out = (float*)d_out;

  const int E = in_sizes[1];
  const int n = in_sizes[0] / NF;

  char* ws = (char*)d_ws;
  size_t off = 0;
  auto alloc = [&](size_t bytes) -> void* {
    void* p = ws + off;
    off = (off + bytes + 255) & ~(size_t)255;
    return p;
  };
  float* s_buf = (float*)alloc((size_t)n * NF * sizeof(float));
  float* h1    = (float*)alloc((size_t)n * NF * sizeof(float));
  float* deg   = (float*)alloc((size_t)n * sizeof(float));
  bf16*  Whi   = (bf16*)alloc((size_t)4 * 16384 * sizeof(bf16));
  bf16*  Wlo   = (bf16*)alloc((size_t)4 * 16384 * sizeof(bf16));

  const int scatter_blocks = (int)(((long long)E * 32 + 255) / 256);
  const int gemm_blocks = (n + 15) / 16;

  // degrees (shared by both layers) + weight bf16 hi/lo split
  hipMemsetAsync(deg, 0, (size_t)n * sizeof(float), stream);
  k_deg<<<(E + 255) / 256, 256, 0, stream>>>(dst, deg, E);
  k_wsplit<<<(4 * 16384) / 256, 256, 0, stream>>>(W_self1, W_neigh1, W_self2, W_neigh2, Whi, Wlo);

  // ---- layer 1 ----
  hipMemsetAsync(s_buf, 0, (size_t)n * NF * sizeof(float), stream);
  k_scatter<<<scatter_blocks, 256, 0, stream>>>(in_feat, weights, src, dst, s_buf, E);
  k_sage_gemm<true><<<gemm_blocks, 256, 0, stream>>>(
      in_feat, s_buf, deg,
      Whi + 0 * 16384, Wlo + 0 * 16384,   // W_self1 hi/lo
      Whi + 1 * 16384, Wlo + 1 * 16384,   // W_neigh1 hi/lo
      b_self1, h1, n);

  // ---- layer 2 ----
  hipMemsetAsync(s_buf, 0, (size_t)n * NF * sizeof(float), stream);
  k_scatter<<<scatter_blocks, 256, 0, stream>>>(h1, weights, src, dst, s_buf, E);
  k_sage_gemm<false><<<gemm_blocks, 256, 0, stream>>>(
      h1, s_buf, deg,
      Whi + 2 * 16384, Wlo + 2 * 16384,   // W_self2 hi/lo
      Whi + 3 * 16384, Wlo + 3 * 16384,   // W_neigh2 hi/lo
      b_self2, out, n);
}
